// RowParallelLinearWithDelta_70669391889075
// MI455X (gfx1250) — compile-verified, matrix operand order checked by reading, not run on hardware
//
#include <hip/hip_runtime.h>
#include <hip/hip_bf16.h>

// ---------------------------------------------------------------------------
// GPTQ dual-adapter GEMM for MI455X (gfx1250, wave32, WMMA).
//   out = x @ (W_base + W_delta) + bias     (weights fused => one GEMM)
// Pass 1: dequantize base+delta 4-bit weights -> combined f16 Wt in [N,K]
//         layout (workspace, ~90 MB).
// Pass 2: tiled WMMA GEMM, f16 inputs, f32 accumulate, bias fused.
//         Block tile 128(M) x 256(N), 8 waves, each wave 64x64 (4x4 WMMA).
//         B tile moved by async global->LDS DMA (ASYNCcnt).
// ---------------------------------------------------------------------------

typedef __attribute__((ext_vector_type(16))) _Float16 v16h;
typedef __attribute__((ext_vector_type(8)))  _Float16 v8h;
typedef __attribute__((ext_vector_type(4)))  _Float16 v4h;
typedef __attribute__((ext_vector_type(8)))  float    v8f;
typedef __attribute__((ext_vector_type(4)))  int      v4i;

#define T_DIM 8192
#define K_DIM 11008
#define N_DIM 4096
#define KP_DIM (K_DIM / 8)   // 1376 packed words along K
#define NW_DIM (N_DIM / 8)   // 512 packed words along N (qzeros)

#define BM 128
#define BN 256
#define BK 32
#define NK (K_DIM / BK)      // 344 k-steps
#define LDA 40               // LDS row stride in halves (32 data + 8 pad, 16B aligned)

#if defined(__has_builtin)
#if __has_builtin(__builtin_amdgcn_global_load_async_to_lds_b128)
#define USE_ASYNC_LDS 1
#endif
#endif

__device__ __forceinline__ void wait_asynccnt0() {
#if defined(__has_builtin) && __has_builtin(__builtin_amdgcn_s_wait_asynccnt)
    __builtin_amdgcn_s_wait_asynccnt(0);
#else
    asm volatile("s_wait_asynccnt 0x0" ::: "memory");
#endif
}

// ---------------------------------------------------------------------------
// Pass 1: combined dequant.  w[k,n] = (qb - (zb+1))*sb + (qd - (zd+1))*sd
// Output layout: Wt[n*K + k]  (f16, n-major => GEMM B tiles read K-contiguous)
// ---------------------------------------------------------------------------
__global__ __launch_bounds__(256)
void gptq_dequant_combined(const int* __restrict__ qw,  const int* __restrict__ qz,
                           const float* __restrict__ sc, const int* __restrict__ gi,
                           const int* __restrict__ dqw, const int* __restrict__ dqz,
                           const float* __restrict__ dsc, const int* __restrict__ dgi,
                           _Float16* __restrict__ Wt)
{
    const int kp = blockIdx.x * 256 + threadIdx.x;
    const int n  = blockIdx.y;
    if (kp >= KP_DIM) return;

    const unsigned wb = (unsigned)qw [(size_t)kp * N_DIM + n];
    const unsigned wd = (unsigned)dqw[(size_t)kp * N_DIM + n];
    const int nWord = n >> 3;
    const int nSh   = (n & 7) * 4;

    v8h h;
#pragma unroll
    for (int j = 0; j < 8; ++j) {
        const int k  = kp * 8 + j;
        const int g  = gi [k];
        const int gd = dgi[k];
        const int qb = (int)((wb >> (4 * j)) & 0xF);
        const int qd = (int)((wd >> (4 * j)) & 0xF);
        const int zb = (int)(((unsigned)qz [(size_t)g  * NW_DIM + nWord] >> nSh) & 0xF) + 1;
        const int zd = (int)(((unsigned)dqz[(size_t)gd * NW_DIM + nWord] >> nSh) & 0xF) + 1;
        const float v = (float)(qb - zb) * sc [(size_t)g  * N_DIM + n]
                      + (float)(qd - zd) * dsc[(size_t)gd * N_DIM + n];
        h[j] = (_Float16)v;
    }
    *(v8h*)(Wt + (size_t)n * K_DIM + (size_t)kp * 8) = h;   // 16B aligned, coalesced
}

// ---------------------------------------------------------------------------
// Pass 2: C[T,N] = Xf32[T,K] (cvt f16) @ Wt[N,K]^T + bias, WMMA f16->f32.
// 256 threads = 8 waves; wave grid 2(M) x 4(N); each wave: 4x4 16x16 tiles
// (64x64 per wave => 16 WMMA per 8 LDS fragments per k-step).
// __launch_bounds__(256, 1): full VGPR budget, no scratch spills.
// ---------------------------------------------------------------------------
__global__ __launch_bounds__(256, 1)
void gptq_wmma_gemm(const float* __restrict__ x, const _Float16* __restrict__ Wt,
                    const float* __restrict__ bias, float* __restrict__ out)
{
    __shared__ _Float16 As[2][BM * LDA];   // 2 * 128 * 40 * 2B = 20 KB
    __shared__ _Float16 Bs[2][BN * LDA];   // 2 * 256 * 40 * 2B = 40 KB ([n][k])

    const int tid  = threadIdx.x;
    const int m0   = blockIdx.y * BM;
    const int n0   = blockIdx.x * BN;
    const int wave = tid >> 5;
    const int lane = tid & 31;
    const int l15  = lane & 15;
    const int hsel = lane >> 4;            // ISA A/B 16-bit layout: K-half select
    const int wm   = wave & 1;             // 2 waves along M (2*64 = 128)
    const int wn   = wave >> 1;            // 4 waves along N (4*64 = 256)

    v8f acc[4][4];
#pragma unroll
    for (int i = 0; i < 4; ++i)
#pragma unroll
        for (int j = 0; j < 4; ++j)
#pragma unroll
            for (int e = 0; e < 8; ++e) acc[i][j][e] = 0.0f;

    float4 aReg[4];   // A: 128x32 f32 tile staged (4 float4 per thread)
#if !defined(USE_ASYNC_LDS)
    uint4  bReg[4];   // B fallback staging
#endif

    auto loadGlobalA = [&](int k0) {
#pragma unroll
        for (int p = 0; p < 4; ++p) {
            const int q = tid + p * 256;          // 1024 float4 chunks
            const int r = q >> 3, c = q & 7;
            aReg[p] = *(const float4*)(x + (size_t)(m0 + r) * K_DIM + k0 + c * 4);
        }
    };

    auto storeLdsA = [&](int buf) {
#pragma unroll
        for (int p = 0; p < 4; ++p) {
            const int q = tid + p * 256;
            const int r = q >> 3, c = q & 7;
            v4h h;
            h[0] = (_Float16)aReg[p].x; h[1] = (_Float16)aReg[p].y;
            h[2] = (_Float16)aReg[p].z; h[3] = (_Float16)aReg[p].w;
            *(v4h*)(&As[buf][r * LDA + c * 4]) = h;      // ds_store_b64
        }
    };

    // B tile: 256 rows x 32 halves = 1024 x 16B chunks, 4 per thread.
#if defined(USE_ASYNC_LDS)
    auto moveB = [&](int k0, int buf) {
#pragma unroll
        for (int p = 0; p < 4; ++p) {
            const int q = tid + p * 256;
            const int j = q >> 2, c = q & 3;
            const _Float16* gp = Wt + (size_t)(n0 + j) * K_DIM + k0 + c * 8;
            _Float16* lp = &Bs[buf][j * LDA + c * 8];
            __builtin_amdgcn_global_load_async_to_lds_b128(
                (__attribute__((address_space(1))) v4i*)(v4i*)(gp),
                (__attribute__((address_space(3))) v4i*)(v4i*)(lp),
                /*offset=*/0, /*cpol=*/0);
        }
    };
#else
    auto loadGlobalB = [&](int k0) {
#pragma unroll
        for (int p = 0; p < 4; ++p) {
            const int q = tid + p * 256;
            const int j = q >> 2, c = q & 3;
            bReg[p] = *(const uint4*)(Wt + (size_t)(n0 + j) * K_DIM + k0 + c * 8);
        }
    };
    auto storeLdsB = [&](int buf) {
#pragma unroll
        for (int p = 0; p < 4; ++p) {
            const int q = tid + p * 256;
            const int j = q >> 2, c = q & 3;
            *(uint4*)(&Bs[buf][j * LDA + c * 8]) = bReg[p];
        }
    };
#endif

    auto compute = [&](int buf) {
        v16h af[4], bf[4];
        // A fragment: lane<16 -> M=l15, K{0..7,16..23}; lane>=16 -> K{8..15,24..31}
#pragma unroll
        for (int tm = 0; tm < 4; ++tm) {
            const int m = wm * 64 + tm * 16 + l15;
            const v8h lo = *(const v8h*)(&As[buf][m * LDA + hsel * 8]);
            const v8h hi = *(const v8h*)(&As[buf][m * LDA + hsel * 8 + 16]);
            af[tm] = __builtin_shufflevector(lo, hi, 0,1,2,3,4,5,6,7,8,9,10,11,12,13,14,15);
        }
#pragma unroll
        for (int tn = 0; tn < 4; ++tn) {
            const int n = wn * 64 + tn * 16 + l15;
            const v8h lo = *(const v8h*)(&Bs[buf][n * LDA + hsel * 8]);
            const v8h hi = *(const v8h*)(&Bs[buf][n * LDA + hsel * 8 + 16]);
            bf[tn] = __builtin_shufflevector(lo, hi, 0,1,2,3,4,5,6,7,8,9,10,11,12,13,14,15);
        }
#pragma unroll
        for (int tm = 0; tm < 4; ++tm)
#pragma unroll
            for (int tn = 0; tn < 4; ++tn)
                acc[tm][tn] = __builtin_amdgcn_wmma_f32_16x16x32_f16(
                    /*neg_a=*/false, af[tm], /*neg_b=*/false, bf[tn],
                    /*c_mod=*/(short)0, acc[tm][tn],
                    /*reuse_a=*/false, /*reuse_b=*/false);
    };

    // --- prologue ---
    loadGlobalA(0);
#if defined(USE_ASYNC_LDS)
    moveB(0, 0);
#else
    loadGlobalB(0);
#endif
    storeLdsA(0);
#if defined(USE_ASYNC_LDS)
    wait_asynccnt0();
#else
    storeLdsB(0);
#endif
    __syncthreads();

    // --- pipelined main loop (double-buffered LDS) ---
    for (int kt = 0; kt < NK; ++kt) {
        const int cur = kt & 1;
        const int nxt = cur ^ 1;
        if (kt + 1 < NK) {
            loadGlobalA((kt + 1) * BK);
#if defined(USE_ASYNC_LDS)
            moveB((kt + 1) * BK, nxt);   // DMA straight into next buffer
#else
            loadGlobalB((kt + 1) * BK);
#endif
        }
        compute(cur);
        if (kt + 1 < NK) {
            storeLdsA(nxt);
#if defined(USE_ASYNC_LDS)
            wait_asynccnt0();
#else
            storeLdsB(nxt);
#endif
        }
        __syncthreads();
    }

    // --- epilogue: C/D layout (VGPR r: lane<16 -> M=r, lane>=16 -> M=r+8) ---
#pragma unroll
    for (int tm = 0; tm < 4; ++tm) {
#pragma unroll
        for (int tn = 0; tn < 4; ++tn) {
            const int row0 = m0 + wm * 64 + tm * 16 + hsel * 8;
            const int col  = n0 + wn * 64 + tn * 16 + l15;
            const float bv = bias[col];
#pragma unroll
            for (int r = 0; r < 8; ++r)
                out[(size_t)(row0 + r) * N_DIM + col] = acc[tm][tn][r] + bv;
        }
    }
}

// ---------------------------------------------------------------------------
extern "C" void kernel_launch(void* const* d_in, const int* in_sizes, int n_in,
                              void* d_out, int out_size, void* d_ws, size_t ws_size,
                              hipStream_t stream) {
    const float* x         = (const float*)d_in[0];
    const int*   qweight   = (const int*)  d_in[1];
    const int*   qzeros    = (const int*)  d_in[2];
    const float* scales    = (const float*)d_in[3];
    const int*   g_idx     = (const int*)  d_in[4];
    const float* bias      = (const float*)d_in[5];
    const int*   d_qweight = (const int*)  d_in[6];
    const int*   d_qzeros  = (const int*)  d_in[7];
    const float* d_scales  = (const float*)d_in[8];
    const int*   d_g_idx   = (const int*)  d_in[9];
    float*       out       = (float*)d_out;

    _Float16* Wt = (_Float16*)d_ws;   // N_DIM * K_DIM * 2 bytes (~90 MB)

    dim3 gD((KP_DIM + 255) / 256, N_DIM);
    gptq_dequant_combined<<<gD, 256, 0, stream>>>(
        qweight, qzeros, scales, g_idx,
        d_qweight, d_qzeros, d_scales, d_g_idx, Wt);

    dim3 gG(N_DIM / BN, T_DIM / BM);   // 16 x 64 blocks
    gptq_wmma_gemm<<<gG, 256, 0, stream>>>(x, Wt, bias, out);
}
